// DirMPNNLayer_23003844837405
// MI455X (gfx1250) — matrix operand order repaired and checked by reading, compiled, not run on hardware
//
#include <hip/hip_runtime.h>
#include <hip/hip_bf16.h>

typedef __attribute__((ext_vector_type(16))) __bf16          v16bf;
typedef __attribute__((ext_vector_type(16))) unsigned short  v16us;
typedef __attribute__((ext_vector_type(8)))  float           v8f;

#define ROWP 136   // LDS row pitch (halves) for a 16x128 tile, padded

// ---------- small helpers ----------

__device__ __forceinline__ unsigned short f2bf(float f) {
  union { float f; unsigned int u; } c; c.f = f;
  unsigned int u = c.u;
  u += 0x7FFFu + ((u >> 16) & 1u);   // round-to-nearest-even
  return (unsigned short)(u >> 16);
}

__device__ __forceinline__ v8f v8zero() {
  v8f z;
#pragma unroll
  for (int i = 0; i < 8; ++i) z[i] = 0.0f;
  return z;
}

__device__ __forceinline__ void atomicAddF(float* p, float v) {
  __hip_atomic_fetch_add(p, v, __ATOMIC_RELAXED, __HIP_MEMORY_SCOPE_AGENT);
}

__device__ __forceinline__ v8f wmma_bf16(v16bf a, v16bf b, v8f c) {
  return __builtin_amdgcn_wmma_f32_16x16x32_bf16(false, a, false, b, (short)0, c,
                                                 false, false);
}

// branchless fetch of 8 consecutive bf16 features (one 16B load) at feature
// index f (f % 8 == 0) from exact concat [s0(l0) | s1(l01-l0) | s2(rest)]
__device__ __forceinline__ uint4 fetch8bf3(int f,
                                           const unsigned short* s0,
                                           const unsigned short* s1,
                                           const unsigned short* s2,
                                           int l0, int l01) {
  const unsigned short* p = s0 + f;
  if (f >= l0)  p = s1 + (f - l0);    // -> v_cndmask on pointer
  if (f >= l01) p = s2 + (f - l01);   // -> v_cndmask on pointer
  return *(const uint4*)p;            // unconditional global_load_b128
}

// 16xK row tile (rows = per-lane concat bf16 features, exact length KC*32)
// times Wt[128][Kpad] -> acc[8] (16x128)
// A layout (16-bit 16x32): lane<16: K = 0..15 ; lane>=16: K = 16..31
// B layout: lane<16: K = 0..15 ; lane>=16: K = 16..31 ; N = lane&15
template<int KC>
__device__ __forceinline__ void gemm_tile(v8f acc[8],
    const unsigned short* s0, const unsigned short* s1, const unsigned short* s2,
    int l0, int l01,
    const unsigned short* __restrict__ Wt, int Kpad, int lane) {
  int m     = lane & 15;
  int koffA = (lane & 16) ? 16 : 0;
  int koffB = (lane & 16) ? 16 : 0;
  const unsigned short* wBase = Wt + (size_t)m * Kpad + koffB;
#pragma unroll 1
  for (int kc = 0; kc < KC; ++kc) {
    int f0 = kc * 32 + koffA;
    union { uint4 q[2]; v16us v; } ua;
    ua.q[0] = fetch8bf3(f0,      s0, s1, s2, l0, l01);
    ua.q[1] = fetch8bf3(f0 + 8,  s0, s1, s2, l0, l01);
    v16bf a = __builtin_bit_cast(v16bf, ua.v);
    v16us bu[8];                       // distinct regs: loads burst, then WMMAs
#pragma unroll
    for (int j = 0; j < 8; ++j)
      bu[j] = *(const v16us*)(wBase + (size_t)j * 16 * Kpad + kc * 32);
#pragma unroll
    for (int j = 0; j < 8; ++j)
      acc[j] = wmma_bf16(a, __builtin_bit_cast(v16bf, bu[j]), acc[j]);
  }
}

// same, but A rows come from a bf16 LDS tile (row pitch ROWP halves), K = 128
__device__ __forceinline__ void gemm_tile_lds(v8f acc[8],
    const unsigned short* aRow, const unsigned short* __restrict__ Wt, int lane) {
  int m     = lane & 15;
  int koffA = (lane & 16) ? 16 : 0;
  int koffB = (lane & 16) ? 16 : 0;
  const unsigned short* wBase = Wt + (size_t)m * 128 + koffB;
#pragma unroll 1
  for (int kc = 0; kc < 4; ++kc) {
    int f0 = kc * 32 + koffA;
    union { uint4 q[2]; v16us v; } ua;
    ua.q[0] = *(const uint4*)(aRow + f0);
    ua.q[1] = *(const uint4*)(aRow + f0 + 8);
    v16bf a = __builtin_bit_cast(v16bf, ua.v);
    v16us bu[8];
#pragma unroll
    for (int j = 0; j < 8; ++j)
      bu[j] = *(const v16us*)(wBase + (size_t)j * 16 * 128 + kc * 32);
#pragma unroll
    for (int j = 0; j < 8; ++j)
      acc[j] = wmma_bf16(a, __builtin_bit_cast(v16bf, bu[j]), acc[j]);
  }
}

// message MLP up to z2 (pre-bias): z1 = [hs|hd|ea]W1+b1 ; h1 = relu(bn1(z1)) ; z2 = h1 W2
__device__ __forceinline__ void msg_z2(v8f acc[8],
    const unsigned short* hs, const unsigned short* hd, const unsigned short* eap,
    const unsigned short* W1t, const unsigned short* W2t,
    const float* b1, const float* g1, const float* be1,
    const float* setL1, unsigned short* ldsTile, int lane) {
  int m  = lane & 15;
  int hi = (lane & 16) ? 8 : 0;
  v8f a1[8];
#pragma unroll
  for (int j = 0; j < 8; ++j) a1[j] = v8zero();
  gemm_tile<9>(a1, hs, hd, eap, 128, 256, W1t, 288, lane);
#pragma unroll
  for (int j = 0; j < 8; ++j) {
    int nn = j * 16 + m;
    float bz = b1[nn], mv = setL1[256 + nn], rv = setL1[384 + nn];
    float gv = g1[nn], bev = be1[nn];
#pragma unroll
    for (int v = 0; v < 8; ++v) {
      float z  = a1[j][v] + bz;
      float hh = fmaxf(0.0f, gv * (z - mv) * rv + bev);
      ldsTile[(size_t)(v + hi) * ROWP + nn] = f2bf(hh);
    }
  }
  asm volatile("s_wait_dscnt 0" ::: "memory");
#pragma unroll
  for (int j = 0; j < 8; ++j) acc[j] = v8zero();
  gemm_tile_lds(acc, ldsTile + (size_t)m * ROWP, W2t, lane);
}

// update MLP up to z2 (pre-bias): z1 = [h|aSt|aTs]U1+b1 ; h1 = relu(bn(z1)) ; z2 = h1 U2
__device__ __forceinline__ void upd_z2(v8f acc[8],
    const unsigned short* p0, const unsigned short* p1, const unsigned short* p2,
    const unsigned short* U1t, const unsigned short* U2t,
    const float* b1, const float* g1, const float* be1,
    const float* setL1, unsigned short* ldsTile, int lane) {
  int m  = lane & 15;
  int hi = (lane & 16) ? 8 : 0;
  v8f a1[8];
#pragma unroll
  for (int j = 0; j < 8; ++j) a1[j] = v8zero();
  gemm_tile<12>(a1, p0, p1, p2, 128, 256, U1t, 384, lane);
#pragma unroll
  for (int j = 0; j < 8; ++j) {
    int nn = j * 16 + m;
    float bz = b1[nn], mv = setL1[256 + nn], rv = setL1[384 + nn];
    float gv = g1[nn], bev = be1[nn];
#pragma unroll
    for (int v = 0; v < 8; ++v) {
      float z  = a1[j][v] + bz;
      float hh = fmaxf(0.0f, gv * (z - mv) * rv + bev);
      ldsTile[(size_t)(v + hi) * ROWP + nn] = f2bf(hh);
    }
  }
  asm volatile("s_wait_dscnt 0" ::: "memory");
#pragma unroll
  for (int j = 0; j < 8; ++j) acc[j] = v8zero();
  gemm_tile_lds(acc, ldsTile + (size_t)m * ROWP, U2t, lane);
}

// ---------- utility kernels ----------

__global__ void k_zero(float* __restrict__ p, size_t cnt) {
  size_t stride = (size_t)gridDim.x * blockDim.x;
  for (size_t i = (size_t)blockIdx.x * blockDim.x + threadIdx.x; i < cnt; i += stride)
    p[i] = 0.0f;
}

// elementwise f32 -> bf16
__global__ void k_cvt(const float* __restrict__ src, unsigned short* __restrict__ dst,
                      size_t cnt) {
  size_t stride = (size_t)gridDim.x * blockDim.x;
  for (size_t i = (size_t)blockIdx.x * blockDim.x + threadIdx.x; i < cnt; i += stride)
    dst[i] = f2bf(src[i]);
}

// f32 [rows][cin] -> bf16 [rows][cout] zero-padded columns
__global__ void k_cvt_pad(const float* __restrict__ src, unsigned short* __restrict__ dst,
                          size_t rows, int cin, int cout) {
  size_t total  = rows * (size_t)cout;
  size_t stride = (size_t)gridDim.x * blockDim.x;
  for (size_t i = (size_t)blockIdx.x * blockDim.x + threadIdx.x; i < total; i += stride) {
    size_t r = i / cout; int c = (int)(i % cout);
    dst[i] = (c < cin) ? f2bf(src[r * cin + c]) : (unsigned short)0;
  }
}

// W is [Kin][128] f32 row-major -> Wt is [128][Kpad] bf16 (zero padded K)
__global__ void k_prep(const float* __restrict__ W, unsigned short* __restrict__ Wt,
                       int Kin, int Kpad) {
  int total = 128 * Kpad;
  for (int i = blockIdx.x * blockDim.x + threadIdx.x; i < total;
       i += gridDim.x * blockDim.x) {
    int nn = i / Kpad, k = i % Kpad;
    float v = (k < Kin) ? W[(size_t)k * 128 + nn] : 0.0f;
    Wt[(size_t)nn * Kpad + k] = f2bf(v);
  }
}

// stat set layout: [sum(128) | sumsq(128) | mean(128) | rstd(128)]
__global__ void k_finalize(float* __restrict__ set, float count) {
  int nn = threadIdx.x;
  if (nn < 128) {
    float s = set[nn], q = set[128 + nn];
    float mean = s / count;
    float var  = q / count - mean * mean;
    set[256 + nn] = mean;
    set[384 + nn] = rsqrtf(var + 1e-5f);
  }
}

// ---------- edge kernels ----------

__global__ __launch_bounds__(128) void k_edge_l1_stats(
    const unsigned short* __restrict__ hb, const int* __restrict__ ei,
    const unsigned short* __restrict__ eab,
    const unsigned short* __restrict__ W1t, const float* __restrict__ b1,
    float* __restrict__ stats, int e, int ntiles) {
  int lane = threadIdx.x & 31, w = threadIdx.x >> 5;
  int wid = blockIdx.x * 4 + w, wstr = gridDim.x * 4;
  int m = lane & 15;
  float s0[8] = {}, q0[8] = {}, s1[8] = {}, q1[8] = {};
  for (int t = wid; t < ntiles; t += wstr) {
    int eidx = t * 16 + m;
    int si = ei[eidx], di = ei[e + eidx];
    const unsigned short* hs  = hb  + (size_t)si * 128;
    const unsigned short* hd  = hb  + (size_t)di * 128;
    const unsigned short* eap = eab + (size_t)eidx * 32;
    {
      v8f acc[8];
#pragma unroll
      for (int j = 0; j < 8; ++j) acc[j] = v8zero();
      gemm_tile<9>(acc, hs, hd, eap, 128, 256, W1t, 288, lane);
#pragma unroll
      for (int j = 0; j < 8; ++j) {
        float bz = b1[j * 16 + m];
#pragma unroll
        for (int v = 0; v < 8; ++v) { float z = acc[j][v] + bz; s0[j] += z; q0[j] += z * z; }
      }
    }
    {
      v8f acc[8];
#pragma unroll
      for (int j = 0; j < 8; ++j) acc[j] = v8zero();
      gemm_tile<9>(acc, hd, hs, eap, 128, 256, W1t, 288, lane);
#pragma unroll
      for (int j = 0; j < 8; ++j) {
        float bz = b1[j * 16 + m];
#pragma unroll
        for (int v = 0; v < 8; ++v) { float z = acc[j][v] + bz; s1[j] += z; q1[j] += z * z; }
      }
    }
  }
#pragma unroll
  for (int j = 0; j < 8; ++j) {
    int nn = j * 16 + m;
    atomicAddF(stats + 0 * 512 + nn,       s0[j]);
    atomicAddF(stats + 0 * 512 + 128 + nn, q0[j]);
    atomicAddF(stats + 1 * 512 + nn,       s1[j]);
    atomicAddF(stats + 1 * 512 + 128 + nn, q1[j]);
  }
}

__global__ __launch_bounds__(128) void k_edge_l2_stats(
    const unsigned short* __restrict__ hb, const int* __restrict__ ei,
    const unsigned short* __restrict__ eab,
    const unsigned short* __restrict__ W1t, const unsigned short* __restrict__ W2t,
    const float* __restrict__ b1, const float* __restrict__ g1, const float* __restrict__ be1,
    const float* __restrict__ b2, float* __restrict__ stats, int e, int ntiles) {
  __shared__ unsigned short h1s[4 * 16 * ROWP];
  int lane = threadIdx.x & 31, w = threadIdx.x >> 5;
  unsigned short* lds = &h1s[w * 16 * ROWP];
  int wid = blockIdx.x * 4 + w, wstr = gridDim.x * 4;
  int m = lane & 15;
  float s0[8] = {}, q0[8] = {}, s1[8] = {}, q1[8] = {};
  for (int t = wid; t < ntiles; t += wstr) {
    int eidx = t * 16 + m;
    int si = ei[eidx], di = ei[e + eidx];
    const unsigned short* hs  = hb  + (size_t)si * 128;
    const unsigned short* hd  = hb  + (size_t)di * 128;
    const unsigned short* eap = eab + (size_t)eidx * 32;
    v8f acc[8];
    msg_z2(acc, hs, hd, eap, W1t, W2t, b1, g1, be1, stats + 0 * 512, lds, lane);
#pragma unroll
    for (int j = 0; j < 8; ++j) {
      float bz = b2[j * 16 + m];
#pragma unroll
      for (int v = 0; v < 8; ++v) { float z = acc[j][v] + bz; s0[j] += z; q0[j] += z * z; }
    }
    msg_z2(acc, hd, hs, eap, W1t, W2t, b1, g1, be1, stats + 1 * 512, lds, lane);
#pragma unroll
    for (int j = 0; j < 8; ++j) {
      float bz = b2[j * 16 + m];
#pragma unroll
      for (int v = 0; v < 8; ++v) { float z = acc[j][v] + bz; s1[j] += z; q1[j] += z * z; }
    }
  }
#pragma unroll
  for (int j = 0; j < 8; ++j) {
    int nn = j * 16 + m;
    atomicAddF(stats + 2 * 512 + nn,       s0[j]);
    atomicAddF(stats + 2 * 512 + 128 + nn, q0[j]);
    atomicAddF(stats + 3 * 512 + nn,       s1[j]);
    atomicAddF(stats + 3 * 512 + 128 + nn, q1[j]);
  }
}

__global__ __launch_bounds__(128) void k_edge_scatter(
    const unsigned short* __restrict__ hb, const int* __restrict__ ei,
    const unsigned short* __restrict__ eab,
    const unsigned short* __restrict__ W1t, const unsigned short* __restrict__ W2t,
    const float* __restrict__ b1, const float* __restrict__ g1, const float* __restrict__ be1,
    const float* __restrict__ b2, const float* __restrict__ g2, const float* __restrict__ be2,
    const float* __restrict__ stats, float* __restrict__ aSt, float* __restrict__ aTs,
    int e, int ntiles) {
  __shared__ unsigned short h1s[4 * 16 * ROWP];
  int lane = threadIdx.x & 31, w = threadIdx.x >> 5;
  unsigned short* lds = &h1s[w * 16 * ROWP];
  int wid = blockIdx.x * 4 + w, wstr = gridDim.x * 4;
  int m = lane & 15, hi = (lane & 16) ? 8 : 0;
  for (int t = wid; t < ntiles; t += wstr) {
    int eidx = t * 16 + m;
    int si = ei[eidx], di = ei[e + eidx];
    const unsigned short* hs  = hb  + (size_t)si * 128;
    const unsigned short* hd  = hb  + (size_t)di * 128;
    const unsigned short* eap = eab + (size_t)eidx * 32;
    int sidx[8];   // rows this lane's accumulator covers (scatter target = src, both dirs)
#pragma unroll
    for (int v = 0; v < 8; ++v) sidx[v] = ei[t * 16 + v + hi];
#pragma unroll 1
    for (int dir = 0; dir < 2; ++dir) {
      v8f acc[8];
      if (dir == 0) msg_z2(acc, hs, hd, eap, W1t, W2t, b1, g1, be1, stats + 0 * 512, lds, lane);
      else          msg_z2(acc, hd, hs, eap, W1t, W2t, b1, g1, be1, stats + 1 * 512, lds, lane);
      const float* setL2 = stats + (2 + dir) * 512;
      float* aOut = dir ? aTs : aSt;
#pragma unroll
      for (int j = 0; j < 8; ++j) {
        int nn = j * 16 + m;
        float bz = b2[nn], mv = setL2[256 + nn], rv = setL2[384 + nn];
        float gv = g2[nn], bev = be2[nn];
#pragma unroll
        for (int v = 0; v < 8; ++v) {
          float z  = acc[j][v] + bz;
          float mm = fmaxf(0.0f, gv * (z - mv) * rv + bev);
          atomicAddF(aOut + (size_t)sidx[v] * 128 + nn, 0.5f * mm);  // alpha = 0.5 both dirs
        }
      }
    }
  }
}

// ---------- node kernels ----------

__global__ __launch_bounds__(128) void k_node_l1_stats(
    const unsigned short* __restrict__ hb, const unsigned short* __restrict__ aStb,
    const unsigned short* __restrict__ aTsb,
    const unsigned short* __restrict__ U1t, const float* __restrict__ b1,
    float* __restrict__ stats, int ntiles) {
  int lane = threadIdx.x & 31, w = threadIdx.x >> 5;
  int wid = blockIdx.x * 4 + w, wstr = gridDim.x * 4;
  int m = lane & 15;
  float s[8] = {}, q[8] = {};
  for (int t = wid; t < ntiles; t += wstr) {
    size_t row = (size_t)(t * 16 + m) * 128;
    v8f acc[8];
#pragma unroll
    for (int j = 0; j < 8; ++j) acc[j] = v8zero();
    gemm_tile<12>(acc, hb + row, aStb + row, aTsb + row, 128, 256, U1t, 384, lane);
#pragma unroll
    for (int j = 0; j < 8; ++j) {
      float bz = b1[j * 16 + m];
#pragma unroll
      for (int v = 0; v < 8; ++v) { float z = acc[j][v] + bz; s[j] += z; q[j] += z * z; }
    }
  }
#pragma unroll
  for (int j = 0; j < 8; ++j) {
    int nn = j * 16 + m;
    atomicAddF(stats + 4 * 512 + nn,       s[j]);
    atomicAddF(stats + 4 * 512 + 128 + nn, q[j]);
  }
}

__global__ __launch_bounds__(128) void k_node_l2_stats(
    const unsigned short* __restrict__ hb, const unsigned short* __restrict__ aStb,
    const unsigned short* __restrict__ aTsb,
    const unsigned short* __restrict__ U1t, const unsigned short* __restrict__ U2t,
    const float* __restrict__ b1, const float* __restrict__ g1, const float* __restrict__ be1,
    const float* __restrict__ b2, float* __restrict__ stats, int ntiles) {
  __shared__ unsigned short h1s[4 * 16 * ROWP];
  int lane = threadIdx.x & 31, w = threadIdx.x >> 5;
  unsigned short* lds = &h1s[w * 16 * ROWP];
  int wid = blockIdx.x * 4 + w, wstr = gridDim.x * 4;
  int m = lane & 15;
  float s[8] = {}, q[8] = {};
  for (int t = wid; t < ntiles; t += wstr) {
    size_t row = (size_t)(t * 16 + m) * 128;
    v8f acc[8];
    upd_z2(acc, hb + row, aStb + row, aTsb + row, U1t, U2t, b1, g1, be1,
           stats + 4 * 512, lds, lane);
#pragma unroll
    for (int j = 0; j < 8; ++j) {
      float bz = b2[j * 16 + m];
#pragma unroll
      for (int v = 0; v < 8; ++v) { float z = acc[j][v] + bz; s[j] += z; q[j] += z * z; }
    }
  }
#pragma unroll
  for (int j = 0; j < 8; ++j) {
    int nn = j * 16 + m;
    atomicAddF(stats + 5 * 512 + nn,       s[j]);
    atomicAddF(stats + 5 * 512 + 128 + nn, q[j]);
  }
}

__global__ __launch_bounds__(128) void k_node_out(
    const unsigned short* __restrict__ hb, const unsigned short* __restrict__ aStb,
    const unsigned short* __restrict__ aTsb,
    const unsigned short* __restrict__ U1t, const unsigned short* __restrict__ U2t,
    const float* __restrict__ b1, const float* __restrict__ g1, const float* __restrict__ be1,
    const float* __restrict__ b2, const float* __restrict__ g2, const float* __restrict__ be2,
    const float* __restrict__ stats, float* __restrict__ out, int ntiles) {
  __shared__ unsigned short h1s[4 * 16 * ROWP];
  int lane = threadIdx.x & 31, w = threadIdx.x >> 5;
  unsigned short* lds = &h1s[w * 16 * ROWP];
  int wid = blockIdx.x * 4 + w, wstr = gridDim.x * 4;
  int m = lane & 15, hi = (lane & 16) ? 8 : 0;
  for (int t = wid; t < ntiles; t += wstr) {
    size_t row = (size_t)(t * 16 + m) * 128;
    v8f acc[8];
    upd_z2(acc, hb + row, aStb + row, aTsb + row, U1t, U2t, b1, g1, be1,
           stats + 4 * 512, lds, lane);
    const float* setL2 = stats + 5 * 512;
#pragma unroll
    for (int j = 0; j < 8; ++j) {
      int nn = j * 16 + m;
      float bz = b2[nn], mv = setL2[256 + nn], rv = setL2[384 + nn];
      float gv = g2[nn], bev = be2[nn];
#pragma unroll
      for (int v = 0; v < 8; ++v) {
        float z = acc[j][v] + bz;
        out[(size_t)(t * 16 + v + hi) * 128 + nn] = fmaxf(0.0f, gv * (z - mv) * rv + bev);
      }
    }
  }
}

// ---------- host ----------

extern "C" void kernel_launch(void* const* d_in, const int* in_sizes, int n_in,
                              void* d_out, int out_size, void* d_ws, size_t ws_size,
                              hipStream_t stream) {
  (void)n_in; (void)out_size; (void)ws_size;
  const float* h    = (const float*)d_in[0];
  const int*   ei   = (const int*)d_in[1];
  const float* ea   = (const float*)d_in[2];
  const float* mW1  = (const float*)d_in[3];
  const float* mb1  = (const float*)d_in[4];
  const float* mg1  = (const float*)d_in[5];
  const float* mbe1 = (const float*)d_in[6];
  const float* mW2  = (const float*)d_in[7];
  const float* mb2  = (const float*)d_in[8];
  const float* mg2  = (const float*)d_in[9];
  const float* mbe2 = (const float*)d_in[10];
  const float* uW1  = (const float*)d_in[11];
  const float* ub1  = (const float*)d_in[12];
  const float* ug1  = (const float*)d_in[13];
  const float* ube1 = (const float*)d_in[14];
  const float* uW2  = (const float*)d_in[15];
  const float* ub2  = (const float*)d_in[16];
  const float* ug2  = (const float*)d_in[17];
  const float* ube2 = (const float*)d_in[18];

  int n  = in_sizes[0] / 128;   // nodes
  int e  = in_sizes[2] / 16;    // edges (edge_attr is e x 16)
  int te = e / 16;              // edge tiles (e divisible by 16 here)
  int tn = n / 16;              // node tiles

  char* ws = (char*)d_ws;
  size_t off = 0;
  auto take = [&](size_t bytes) { size_t o = off; off += (bytes + 255) & ~(size_t)255; return o; };
  unsigned short* W1t  = (unsigned short*)(ws + take((size_t)128 * 288 * 2));
  unsigned short* W2t  = (unsigned short*)(ws + take((size_t)128 * 128 * 2));
  unsigned short* U1t  = (unsigned short*)(ws + take((size_t)128 * 384 * 2));
  unsigned short* U2t  = (unsigned short*)(ws + take((size_t)128 * 128 * 2));
  unsigned short* hbf  = (unsigned short*)(ws + take((size_t)n * 128 * 2));
  unsigned short* eabf = (unsigned short*)(ws + take((size_t)e * 32 * 2));  // padded to 32
  unsigned short* aStb = (unsigned short*)(ws + take((size_t)n * 128 * 2));
  unsigned short* aTsb = (unsigned short*)(ws + take((size_t)n * 128 * 2));
  float* stats = (float*)(ws + off); off += (size_t)6 * 512 * 4;       // 6 stat sets
  float* aSt   = (float*)(ws + off); off += (size_t)n * 128 * 4;       // alpha * segsum(m_st)
  float* aTs   = (float*)(ws + off); off += (size_t)n * 128 * 4;       // (1-alpha) * segsum(m_ts)

  // zero stats + fp32 accumulators (contiguous region)
  size_t zcnt = (size_t)6 * 512 + (size_t)2 * n * 128;
  k_zero<<<2048, 256, 0, stream>>>(stats, zcnt);

  // bf16 pre-conversion of activations + weight transpose prep
  k_cvt<<<2048, 256, 0, stream>>>(h, hbf, (size_t)n * 128);
  k_cvt_pad<<<2048, 256, 0, stream>>>(ea, eabf, (size_t)e, 16, 32);
  k_prep<<<(128 * 288 + 255) / 256, 256, 0, stream>>>(mW1, W1t, 272, 288);
  k_prep<<<(128 * 128 + 255) / 256, 256, 0, stream>>>(mW2, W2t, 128, 128);
  k_prep<<<(128 * 384 + 255) / 256, 256, 0, stream>>>(uW1, U1t, 384, 384);
  k_prep<<<(128 * 128 + 255) / 256, 256, 0, stream>>>(uW2, U2t, 128, 128);

  int eb = (te + 3) / 4; if (eb > 4096) eb = 4096; if (eb < 1) eb = 1;
  int nb = (tn + 3) / 4; if (nb > 2048) nb = 2048; if (nb < 1) nb = 1;

  // message MLP: BN1 stats (both directions)
  k_edge_l1_stats<<<eb, 128, 0, stream>>>(hbf, ei, eabf, W1t, mb1, stats, e, te);
  k_finalize<<<1, 128, 0, stream>>>(stats + 0 * 512, (float)e);
  k_finalize<<<1, 128, 0, stream>>>(stats + 1 * 512, (float)e);
  // message MLP: BN2 stats
  k_edge_l2_stats<<<eb, 128, 0, stream>>>(hbf, ei, eabf, W1t, W2t, mb1, mg1, mbe1, mb2,
                                          stats, e, te);
  k_finalize<<<1, 128, 0, stream>>>(stats + 2 * 512, (float)e);
  k_finalize<<<1, 128, 0, stream>>>(stats + 3 * 512, (float)e);
  // message MLP: final + scatter by src
  k_edge_scatter<<<eb, 128, 0, stream>>>(hbf, ei, eabf, W1t, W2t, mb1, mg1, mbe1, mb2,
                                         mg2, mbe2, stats, aSt, aTs, e, te);

  // convert aggregated messages to bf16 for the update MLP
  k_cvt<<<2048, 256, 0, stream>>>(aSt, aStb, (size_t)n * 128);
  k_cvt<<<2048, 256, 0, stream>>>(aTs, aTsb, (size_t)n * 128);

  // update MLP
  k_node_l1_stats<<<nb, 128, 0, stream>>>(hbf, aStb, aTsb, U1t, ub1, stats, tn);
  k_finalize<<<1, 128, 0, stream>>>(stats + 4 * 512, (float)n);
  k_node_l2_stats<<<nb, 128, 0, stream>>>(hbf, aStb, aTsb, U1t, U2t, ub1, ug1, ube1, ub2,
                                          stats, tn);
  k_finalize<<<1, 128, 0, stream>>>(stats + 5 * 512, (float)n);
  k_node_out<<<nb, 128, 0, stream>>>(hbf, aStb, aTsb, U1t, U2t, ub1, ug1, ube1, ub2,
                                     ug2, ube2, stats, (float*)d_out, tn);
}
